// GCN_58110907515029
// MI455X (gfx1250) — compile-verified, matrix operand order checked by reading, not run on hardware
//
#include <hip/hip_runtime.h>
#include <hip/hip_bf16.h>

#define NN   50000
#define EE   800000
#define DIN  512
#define DH   256
#define DOUT 64

typedef __attribute__((ext_vector_type(16))) __bf16 v16bf;
typedef __attribute__((ext_vector_type(8)))  float  v8f;

// ---------- helpers ----------
__device__ inline unsigned int pkbf(float lo, float hi) {
  // pack two f32 -> two bf16 (round-to-nearest-ish), lo in [15:0], hi in [31:16]
  unsigned int ul = __float_as_uint(lo);
  unsigned int uh = __float_as_uint(hi);
  return ((uh + 0x8000u) & 0xFFFF0000u) | ((ul + 0x8000u) >> 16);
}

__device__ inline void atomAddF(float* p, float v) {
  unsafeAtomicAdd(p, v);   // lowers to global_atomic_add_f32 (L2-side FP atomic)
}

// ---------- tiny utility kernels ----------
__global__ void fill_kernel(float* __restrict__ p, float v, long long n) {
  long long i = (long long)blockIdx.x * blockDim.x + threadIdx.x;
  if (i < n) p[i] = v;
}

// row-major f32 -> row-major bf16 (packed pairs); one thread = 8 floats -> uint4
__global__ void pack_bf16_kernel(const float* __restrict__ src,
                                 unsigned int* __restrict__ dst, long long nvec) {
  long long i = (long long)blockIdx.x * blockDim.x + threadIdx.x;
  if (i >= nvec) return;
  const float4 f0 = *(const float4*)(src + i * 8);
  const float4 f1 = *(const float4*)(src + i * 8 + 4);
  uint4 o;
  o.x = pkbf(f0.x, f0.y); o.y = pkbf(f0.z, f0.w);
  o.z = pkbf(f1.x, f1.y); o.w = pkbf(f1.z, f1.w);
  ((uint4*)dst)[i] = o;
}

// WT[n][k] (bf16, packed pairs) from W[k][n] (f32). count = Ncol*K/2 dwords.
__global__ void transpose_pack_kernel(const float* __restrict__ W,
                                      unsigned int* __restrict__ WT,
                                      int Ncol, int K) {
  int gid = blockIdx.x * blockDim.x + threadIdx.x;
  int half = K >> 1;
  if (gid >= Ncol * half) return;
  int n  = gid / half;
  int kk = gid - n * half;
  float a = W[(size_t)(2 * kk)     * Ncol + n];
  float b = W[(size_t)(2 * kk + 1) * Ncol + n];
  WT[gid] = pkbf(a, b);
}

__global__ void deg_kernel(const int* __restrict__ dst, float* __restrict__ deg, int e) {
  int i = blockIdx.x * blockDim.x + threadIdx.x;
  if (i < e) atomAddF(deg + dst[i], 1.0f);
}

__global__ void rsqrt_kernel(float* __restrict__ d, int n) {
  int i = blockIdx.x * blockDim.x + threadIdx.x;
  if (i < n) d[i] = rsqrtf(d[i]);   // deg >= 1 always (self loop)
}

// ---------- WMMA GEMM: C = A(bf16)[M x K] * BT(bf16)[Ncol x K] ----------
// One wave computes a 16x(CT*16) strip: A fragment reused CT times per k-step.
// Per-lane operand layout (ISA 7.12.2, 16-bit 16x32):
//   lane&15 = row/col, lane<16 -> K 0..7 & 16..23, lane>=16 -> K 8..15 & 24..31.
template<int K, int LDO, int CT>
__global__ void gemm_wmma_kernel(const unsigned int* __restrict__ A,
                                 const unsigned int* __restrict__ BT,
                                 float* __restrict__ C, int Mtiles) {
  const int lane  = threadIdx.x;                              // 0..31
  const int tileM = blockIdx.x * blockDim.y + threadIdx.y;
  if (tileM >= Mtiles) return;
  const int n0    = blockIdx.y * (CT * 16);
  const int r     = tileM * 16 + (lane & 15);
  const int khalf = (lane >> 4) * 8;

  const unsigned int* arow = A + (size_t)r * (K / 2);
  const unsigned int* brow[CT];
#pragma unroll
  for (int t = 0; t < CT; ++t)
    brow[t] = BT + (size_t)(n0 + t * 16 + (lane & 15)) * (K / 2);

  v8f acc[CT];
#pragma unroll
  for (int t = 0; t < CT; ++t) acc[t] = (v8f){};

#pragma unroll 4
  for (int k0 = 0; k0 < K; k0 += 32) {
    union { uint4 q[2]; v16bf v; } a;
    a.q[0] = *(const uint4*)(arow + ((k0 + khalf) >> 1));
    a.q[1] = *(const uint4*)(arow + ((k0 + 16 + khalf) >> 1));
#pragma unroll
    for (int t = 0; t < CT; ++t) {
      union { uint4 q[2]; v16bf v; } b;
      b.q[0] = *(const uint4*)(brow[t] + ((k0 + khalf) >> 1));
      b.q[1] = *(const uint4*)(brow[t] + ((k0 + 16 + khalf) >> 1));
      acc[t] = __builtin_amdgcn_wmma_f32_16x16x32_bf16(
          false, a.v, false, b.v, (short)0, acc[t], false, false);
    }
  }

  // C/D layout: lanes 0-15 -> M = j, lanes 16-31 -> M = 8+j; N = lane&15
  const int mofs = (lane < 16) ? 0 : 8;
#pragma unroll
  for (int t = 0; t < CT; ++t) {
    float* crow = C + (size_t)(tileM * 16) * LDO + n0 + t * 16 + (lane & 15);
#pragma unroll
    for (int j = 0; j < 8; ++j)
      crow[(size_t)(mofs + j) * LDO] = acc[t][j];
  }
}

// ---------- scatter aggregation: agg[dst] += h[src] * dinv[src]*dinv[dst] ----------
// One wave per (edge or self-loop); lanes cover D floats interleaved (coalesced 128B).
template<int D>
__global__ void agg_kernel(const int* __restrict__ ei, const float* __restrict__ dinv,
                           const float* __restrict__ h, float* __restrict__ agg,
                           int e, int n) {
  int wave = (blockIdx.x * blockDim.x + threadIdx.x) >> 5;
  int lane = threadIdx.x & 31;
  int total = e + n;
  if (wave >= total) return;
  int s, d;
  if (wave < e) { s = ei[wave]; d = ei[e + wave]; }
  else          { s = d = wave - e; }
  // wave-uniform: force scalar so address math goes SALU / s_load
  s = __builtin_amdgcn_readfirstlane(s);
  d = __builtin_amdgcn_readfirstlane(d);
  float c = dinv[s] * dinv[d];
  const float* hs = h + (size_t)s * D;
  float* ad = agg + (size_t)d * D;
#pragma unroll
  for (int j = 0; j < D / 32; ++j) {
    int i = lane + j * 32;
    atomAddF(ad + i, hs[i] * c);
  }
}

// ---------- bias + ReLU + bf16 pack: h1bf[r][k pairs] ----------
__global__ void relu_bias_pack_kernel(const float* __restrict__ agg,
                                      const float* __restrict__ b1,
                                      unsigned int* __restrict__ h1bf) {
  long long gid = (long long)blockIdx.x * blockDim.x + threadIdx.x;
  const int half = DH / 2;
  if (gid >= (long long)NN * half) return;
  int row = (int)(gid / half);
  int p   = (int)(gid - (long long)row * half);
  int d0  = 2 * p;
  float v0 = fmaxf(agg[(size_t)row * DH + d0]     + b1[d0],     0.0f);
  float v1 = fmaxf(agg[(size_t)row * DH + d0 + 1] + b1[d0 + 1], 0.0f);
  h1bf[gid] = pkbf(v0, v1);
}

// ---------- fused bias + log_softmax + argmax (one wave per row of 64) ----------
__global__ void finalize_kernel(const float* __restrict__ agg2,
                                const float* __restrict__ b2,
                                float* __restrict__ out) {
  int row  = blockIdx.x * (blockDim.x >> 5) + (threadIdx.x >> 5);
  int lane = threadIdx.x & 31;
  if (row >= NN) return;
  int c0 = 2 * lane, c1 = 2 * lane + 1;
  float2 a = *(const float2*)(agg2 + (size_t)row * DOUT + c0);
  float v0 = a.x + b2[c0];
  float v1 = a.y + b2[c1];
  // tuple element 0: pre-softmax logits h
  out[(size_t)row * DOUT + c0] = v0;
  out[(size_t)row * DOUT + c1] = v1;
  // max reduce
  float m = fmaxf(v0, v1);
#pragma unroll
  for (int off = 16; off; off >>= 1) m = fmaxf(m, __shfl_xor(m, off, 32));
  // sum(exp) reduce
  float s = __expf(v0 - m) + __expf(v1 - m);
#pragma unroll
  for (int off = 16; off; off >>= 1) s += __shfl_xor(s, off, 32);
  float lse = __logf(s);
  float* o2 = out + (size_t)NN * DOUT;
  o2[(size_t)row * DOUT + c0] = v0 - m - lse;
  o2[(size_t)row * DOUT + c1] = v1 - m - lse;
  // argmax (first max index)
  float bv; int bi;
  if (v0 >= v1) { bv = v0; bi = c0; } else { bv = v1; bi = c1; }
#pragma unroll
  for (int off = 16; off; off >>= 1) {
    float ov = __shfl_xor(bv, off, 32);
    int   oi = __shfl_xor(bi, off, 32);
    if (ov > bv || (ov == bv && oi < bi)) { bv = ov; bi = oi; }
  }
  if (lane == 0) out[(size_t)2 * NN * DOUT + row] = (float)bi;
}

// ---------- host orchestration ----------
extern "C" void kernel_launch(void* const* d_in, const int* in_sizes, int n_in,
                              void* d_out, int out_size, void* d_ws, size_t ws_size,
                              hipStream_t stream) {
  const float* x  = (const float*)d_in[0];
  const int*   ei = (const int*)d_in[1];      // [2, E]: row0 = src, row1 = dst
  const float* W1 = (const float*)d_in[2];
  const float* b1 = (const float*)d_in[3];
  const float* W2 = (const float*)d_in[4];
  const float* b2 = (const float*)d_in[5];
  float* out = (float*)d_out;

  // workspace layout (floats), ~154 MB total with buffer reuse (L2-resident)
  float* ws   = (float*)d_ws;
  float* dinv = ws;                                             // N
  unsigned int* xbf = (unsigned int*)(dinv + NN);               // N*DIN bf16 = N*DIN/2 dwords
  float* bufH = (float*)(xbf + (size_t)NN * (DIN / 2));         // N*DH  (h1 pre-agg; later h1bf)
  float* bufG = bufH + (size_t)NN * DH;                         // N*DH  (agg1; later h2 + agg2)
  unsigned int* w1t = (unsigned int*)(bufG + (size_t)NN * DH);  // DH*DIN/2 dwords
  unsigned int* w2t = w1t + (DH * DIN / 2);                     // DOUT*DH/2 dwords

  unsigned int* h1bf = (unsigned int*)bufH;                     // N*DH bf16 (reuse)
  float* h2   = bufG;                                           // N*DOUT (reuse)
  float* agg2 = bufG + (size_t)NN * DOUT;                       // N*DOUT (reuse)

  const int T = 256;

  // x -> bf16 (single pass; removes per-tile repacking from the GEMM loop)
  {
    long long nvec = (long long)NN * DIN / 8;
    pack_bf16_kernel<<<(nvec + T - 1) / T, T, 0, stream>>>(x, xbf, nvec);
  }

  // weight transpose+pack to WMMA B layout
  transpose_pack_kernel<<<(DH * DIN / 2 + T - 1) / T, T, 0, stream>>>(W1, w1t, DH, DIN);
  transpose_pack_kernel<<<(DOUT * DH / 2 + T - 1) / T, T, 0, stream>>>(W2, w2t, DOUT, DH);

  // degree (init 1.0 for self loop) -> dinv
  fill_kernel<<<(NN + T - 1) / T, T, 0, stream>>>(dinv, 1.0f, NN);
  deg_kernel<<<(EE + T - 1) / T, T, 0, stream>>>(ei + EE, dinv, EE);
  rsqrt_kernel<<<(NN + T - 1) / T, T, 0, stream>>>(dinv, NN);

  // GEMM1: h = x @ W1 ; 3125 row tiles, 16 col tiles (CT=4 -> 4 strips)
  fill_kernel<<<((long long)NN * DH + T - 1) / T, T, 0, stream>>>(bufG, 0.0f, (long long)NN * DH);
  {
    const int Mtiles = NN / 16;                       // 3125
    dim3 blk(32, 4), grd((Mtiles + 3) / 4, (DH / 16) / 4);
    gemm_wmma_kernel<DIN, DH, 4><<<grd, blk, 0, stream>>>(xbf, w1t, bufH, Mtiles);
  }

  // aggregation layer 1 (E + N self loops), wave per edge
  {
    int waves = EE + NN;
    agg_kernel<DH><<<(waves + 7) / 8, T, 0, stream>>>(ei, dinv, bufH, bufG, EE, NN);
  }

  // bias + relu + bf16 pack (bufG -> h1bf in bufH)
  relu_bias_pack_kernel<<<((long long)NN * (DH / 2) + T - 1) / T, T, 0, stream>>>(bufG, b1, h1bf);

  // GEMM2: h2 = h1 @ W2 ; 4 col tiles in one CT=4 strip
  {
    const int Mtiles = NN / 16;
    dim3 blk(32, 4), grd((Mtiles + 3) / 4, 1);
    gemm_wmma_kernel<DH, DOUT, 4><<<grd, blk, 0, stream>>>(h1bf, w2t, h2, Mtiles);
  }

  // aggregation layer 2
  fill_kernel<<<((long long)NN * DOUT + T - 1) / T, T, 0, stream>>>(agg2, 0.0f, (long long)NN * DOUT);
  {
    int waves = EE + NN;
    agg_kernel<DOUT><<<(waves + 7) / 8, T, 0, stream>>>(ei, dinv, h2, agg2, EE, NN);
  }

  // bias + log_softmax + argmax -> d_out (h | logsoftmax | preds)
  finalize_kernel<<<(NN + 7) / 8, T, 0, stream>>>(agg2, b2, out);
}
